// DMPNet_75900662055053
// MI455X (gfx1250) — matrix operand order repaired and checked by reading, not compiled, not run on hardware
//
#include <hip/hip_runtime.h>
#include <hip/hip_bf16.h>

typedef _Float16 half_t;
typedef __attribute__((ext_vector_type(16))) _Float16 v16h;
typedef __attribute__((ext_vector_type(8)))  _Float16 v8h;
typedef __attribute__((ext_vector_type(8)))  float    v8f;

// ---------------------------------------------------------------------------
// f32 -> f16 conversion helpers (feed WMMA operands)
// ---------------------------------------------------------------------------
__global__ void cvt_f32_f16(const float* __restrict__ src, half_t* __restrict__ dst, int n) {
    int i = blockIdx.x * blockDim.x + threadIdx.x;
    if (i < n) dst[i] = (half_t)src[i];
}

// Wl is (54,2048); pad to (64,2048) so GEMM3 N-tile is a clean 64.
__global__ void pad_wl_f16(const float* __restrict__ wl, half_t* __restrict__ dst) {
    int i = blockIdx.x * blockDim.x + threadIdx.x;      // 64*2048 threads
    int row = i >> 11, c = i & 2047;
    dst[i] = (half_t)((row < 54) ? wl[row * 2048 + c] : 0.0f);
}

__global__ void pad_bl_f32(const float* __restrict__ bl, float* __restrict__ dst) {
    int i = threadIdx.x;
    if (i < 64) dst[i] = (i < 54) ? bl[i] : 0.0f;
}

// ---------------------------------------------------------------------------
// WMMA GEMM:  C[m,n] = epilogue( sum_k A[m,k]*B[n,k] + bias[n] )
// A: (M,K) f16 row-major, B: (N,K) f16 row-major (i.e. W as stored, need W^T)
// One wave -> (MI*16) x 64 output tile: MI A-frags, 4 B-tiles (one B frag live
// at a time), MI*4 f32 accumulators, MI*4 WMMAs per K=32 step.
// MODE 0: tanh -> f16 out      MODE 1: (acc+bias)*100 -> f32 out
// ---------------------------------------------------------------------------
__device__ __forceinline__ v16h load_a_frag(const half_t* __restrict__ A, int lda,
                                            int row, int k0, int hi) {
    // A 16x32 layout (matches B): lanes 0-15 hold K = k0+0..15,
    // lanes 16-31 hold K = k0+16..31  (hi = lane>>4)
    const v8h* p0 = reinterpret_cast<const v8h*>(A + (size_t)row * lda + k0 + hi * 16);
    const v8h* p1 = reinterpret_cast<const v8h*>(A + (size_t)row * lda + k0 + hi * 16 + 8);
    v8h a0 = *p0, a1 = *p1;
    v16h r;
#pragma unroll
    for (int i = 0; i < 8; ++i) { r[i] = a0[i]; r[i + 8] = a1[i]; }
    return r;
}

template <int MI, int MODE>
__global__ __launch_bounds__(256) void wmma_gemm(const half_t* __restrict__ A,
                                                 const half_t* __restrict__ B,
                                                 const float*  __restrict__ bias,
                                                 void* __restrict__ Cout,
                                                 int M, int N, int K) {
    const int wave   = (blockIdx.x * blockDim.x + threadIdx.x) >> 5;
    const int lane   = threadIdx.x & 31;
    const int tilesN = N >> 6;                 // 64-wide N tiles
    const int tm     = wave / tilesN;          // (MI*16)-tall M tiles
    const int tn     = wave - tm * tilesN;
    if (tm * (MI * 16) >= M) return;

    const int col = lane & 15;                 // n (or m) within 16
    const int hi  = lane >> 4;                 // K-half selector

    v8f acc[MI][4] = {};                       // zero-init accumulators

    for (int k0 = 0; k0 < K; k0 += 32) {
        v16h a[MI];
#pragma unroll
        for (int mi = 0; mi < MI; ++mi)
            a[mi] = load_a_frag(A, K, tm * (MI * 16) + mi * 16 + col, k0, hi);
#pragma unroll
        for (int nt = 0; nt < 4; ++nt) {
            // B 32x16 layout: lane = column n, lanes 0-15 K=k0..k0+15,
            // lanes 16-31 K=k0+16..k0+31 -> one contiguous 32B chunk per lane.
            const v16h b = *reinterpret_cast<const v16h*>(
                B + (size_t)(tn * 64 + nt * 16 + col) * K + k0 + hi * 16);
#pragma unroll
            for (int mi = 0; mi < MI; ++mi)
                acc[mi][nt] = __builtin_amdgcn_wmma_f32_16x16x32_f16(
                    false, a[mi], false, b, (short)0, acc[mi][nt], false, false);
        }
    }

#pragma unroll
    for (int mi = 0; mi < MI; ++mi) {
#pragma unroll
        for (int nt = 0; nt < 4; ++nt) {
            const int n  = tn * 64 + nt * 16 + col;
            const float bv = bias[n];
#pragma unroll
            for (int r = 0; r < 8; ++r) {
                const int m = tm * (MI * 16) + mi * 16 + hi * 8 + r;  // C/D layout
                float v = acc[mi][nt][r] + bv;
                if (MODE == 0) {
                    ((half_t*)Cout)[(size_t)m * N + n] = (half_t)tanhf(v);
                } else {
                    ((float*)Cout)[(size_t)m * N + n] = v * 100.0f;
                }
            }
        }
    }
}

// ---------------------------------------------------------------------------
// DMP Euler integration: one thread per row r = b*9+d; 100 steps, sample /10.
// Per-step coefficients g[s][n] = x_s * psi_n(x_s) / sum_psi are identical for
// all rows -> compute once per block into LDS (broadcast reads thereafter).
// ---------------------------------------------------------------------------
__global__ __launch_bounds__(256) void dmp_integrate(const float* __restrict__ out3,  // (B,64)
                                                     const float* __restrict__ input, // (B,128)
                                                     float* __restrict__ out,         // (B,9,10)
                                                     int R) {
    __shared__ float g[100][5];
    {
        const int s = threadIdx.x;
        if (s < 100) {
            const float x = powf(0.99f, (float)(s + 1));   // x_s after s+1 canonical updates
            float psi[5], psum = 0.0f;
#pragma unroll
            for (int n = 0; n < 5; ++n) {
                const float c = expf(-0.25f * (float)n);   // rbf centers
                const float h = 11.180339887f / c;         // widths: 5^1.5 / c
                const float d = x - c;
                psi[n] = expf(-h * d * d);
                psum += psi[n];
            }
            const float inv = x / psum;
#pragma unroll
            for (int n = 0; n < 5; ++n) g[s][n] = psi[n] * inv;
        }
    }
    __syncthreads();

    const int r = blockIdx.x * blockDim.x + threadIdx.x;
    if (r >= R) return;
    const int b = r / 9, d = r - b * 9;

    const float* o = out3 + (size_t)b * 64;
    const float goal = o[d];
    const float w0 = o[9 + d * 5 + 0], w1 = o[9 + d * 5 + 1], w2 = o[9 + d * 5 + 2],
                w3 = o[9 + d * 5 + 3], w4 = o[9 + d * 5 + 4];
    const float y0  = input[(size_t)b * 128 + 7 + d];
    const float dy0 = input[(size_t)b * 128 + 22 + d];
    const float scale = goal - y0;

    float y = y0, z = dy0, prev = y0;
    float* op = out + (size_t)b * 90 + d * 10;

    int s = 0;
    for (int t = 0; t < 10; ++t) {
#pragma unroll
        for (int ss = 0; ss < 10; ++ss) {
            const float* gs = g[s + ss];
            const float fx = scale * (w0 * gs[0] + w1 * gs[1] + w2 * gs[2] +
                                      w3 * gs[3] + w4 * gs[4]);
            const float dz = 25.0f * (6.25f * (goal - y) - z) + fx;   // a_z=25, b_z=6.25
            y += z * 0.01f;                                           // dy = z (old), *DT
            z += dz * 0.01f;
        }
        s += 10;
        op[t] = y - prev;
        prev = y;
    }
}

// ---------------------------------------------------------------------------
// Launch
// ---------------------------------------------------------------------------
extern "C" void kernel_launch(void* const* d_in, const int* in_sizes, int n_in,
                              void* d_out, int out_size, void* d_ws, size_t ws_size,
                              hipStream_t stream) {
    (void)n_in; (void)out_size; (void)ws_size;
    const float* input = (const float*)d_in[0];
    const float* W0    = (const float*)d_in[1];
    const float* b0    = (const float*)d_in[2];
    const float* W1    = (const float*)d_in[3];
    const float* b1    = (const float*)d_in[4];
    const float* Wl    = (const float*)d_in[5];
    const float* bl    = (const float*)d_in[6];

    const int B   = in_sizes[0] / 128;     // 16384
    const int HID = 2048;

    char* ws = (char*)d_ws;
    size_t off = 0;
    half_t* Ah   = (half_t*)(ws + off); off += (size_t)B * 128 * 2;      // input f16
    half_t* W0h  = (half_t*)(ws + off); off += (size_t)HID * 128 * 2;
    half_t* W1h  = (half_t*)(ws + off); off += (size_t)HID * HID * 2;
    half_t* Wlh  = (half_t*)(ws + off); off += (size_t)64 * HID * 2;     // padded
    float*  blp  = (float*)(ws + off);  off += 256;                      // 64 f32
    half_t* h1   = (half_t*)(ws + off); off += (size_t)B * HID * 2;
    half_t* h2   = (half_t*)(ws + off); off += (size_t)B * HID * 2;
    float*  out3 = (float*)(ws + off);  off += (size_t)B * 64 * 4;

    const int T = 256;
    cvt_f32_f16<<<(B * 128 + T - 1) / T, T, 0, stream>>>(input, Ah, B * 128);
    cvt_f32_f16<<<(HID * 128 + T - 1) / T, T, 0, stream>>>(W0, W0h, HID * 128);
    cvt_f32_f16<<<(HID * HID + T - 1) / T, T, 0, stream>>>(W1, W1h, HID * HID);
    pad_wl_f16<<<(64 * HID) / T, T, 0, stream>>>(Wl, Wlh);
    pad_bl_f32<<<1, 64, 0, stream>>>(bl, blp);

    // waves = (M/(MI*16))*(N/64); 8 waves per 256-thread block
    auto gemm_blocks = [](int M, int N, int mi) {
        return ((M / (mi * 16)) * (N / 64) + 7) / 8;
    };

    // GEMM1: K=128 short -> 32x64 tiles (more waves, better latency hiding)
    wmma_gemm<2, 0><<<gemm_blocks(B, HID, 2), T, 0, stream>>>(Ah, W0h, b0, h1, B, HID, 128);
    // GEMM2: dominant, K=2048 -> 64x64 tiles for 32 FLOP/byte at the L2 interface
    wmma_gemm<4, 0><<<gemm_blocks(B, HID, 4), T, 0, stream>>>(h1, W1h, b1, h2, B, HID, HID);
    // GEMM3: N=64 (padded) -> 32x64 tiles
    wmma_gemm<2, 1><<<gemm_blocks(B, 64, 2),  T, 0, stream>>>(h2, Wlh, blp, out3, B, 64, HID);

    const int R = B * 9;
    dmp_integrate<<<(R + T - 1) / T, T, 0, stream>>>(out3, input, (float*)d_out, R);
}
